// MultiHeadAttention_13426067768081
// MI455X (gfx1250) — matrix operand order repaired and checked by reading, not compiled, not run on hardware
//
#include <hip/hip_runtime.h>
#include <hip/hip_fp16.h>

typedef __attribute__((ext_vector_type(16))) _Float16 v16h;
typedef __attribute__((ext_vector_type(8)))  float    v8f;

#define D_MODEL   1024
#define NUM_HEADS 16
#define DEPTH     64
#define BATCH     2
#define SEQ       2048

// ---------------------------------------------------------------------------
// WMMA helpers (gfx1250, wave32). D = A(16x32 f16) * B(32x16 f16) + C(f32)
// ---------------------------------------------------------------------------
__device__ __forceinline__ v8f wmma_f16(v16h a, v16h b, v8f c) {
  // 8 args: (neg_a, A, neg_b, B, c_mod, C, reuse_a, reuse_b)
  return __builtin_amdgcn_wmma_f32_16x16x32_f16(false, a, false, b, (short)0, c,
                                                false, false);
}

// A fragment: 16x32 f16, contiguous K order used by this code:
//   lanes 0-15:  M=lane,    K = 0..15
//   lanes 16-31: M=lane-16, K = 16..31
__device__ __forceinline__ void load_a_frag(const _Float16* lds, int lda,
                                            int mBase, int lane, v16h& a) {
  const _Float16* row = lds + (size_t)(mBase + (lane & 15)) * lda;
  const int koff = (lane >> 4) << 4;  // 0 or 16
#pragma unroll
  for (int j = 0; j < 16; ++j) {
    a[j] = row[koff + j];
  }
}

// B fragment: 32x16 f16 (KxN), contiguous K order matching load_a_frag:
//   lane holds column N = lane%16; lanes 0-15: K=0..15, lanes 16-31: K=16..31
__device__ __forceinline__ void load_b_frag(const _Float16* lds, int ldb,
                                            int nBase, int lane, v16h& b) {
  const int n    = nBase + (lane & 15);
  const int koff = (lane >> 4) << 4;  // 0 or 16
#pragma unroll
  for (int j = 0; j < 16; ++j) {
    b[j] = lds[(size_t)(koff + j) * ldb + n];
  }
}

// Dump both 16x16 f32 accumulators of this wave into a 64x65 LDS tile so the
// epilogue can issue row-contiguous (coalesced, b128-fusable) global stores.
__device__ __forceinline__ void stage_c(float (*Cs)[65], int mt, int nt,
                                        int lane, const v8f& a0, const v8f& a1) {
  const int cn = nt + (lane & 15);
  const int rb = (lane >> 4) << 3;  // 0 or 8
#pragma unroll
  for (int r = 0; r < 8; ++r) {
    Cs[mt + rb + r][cn]      = a0[r];
    Cs[mt + 16 + rb + r][cn] = a1[r];
  }
}

// ---------------------------------------------------------------------------
// Kernel 1: projection  Y[b,h,s,d] = (X @ W + bias) split into heads, f16.
// X f32 [B*S, 1024], W f32 [1024,1024]. 256 thr = 8 waves, 64x64 tile, K=32.
// ---------------------------------------------------------------------------
__global__ __launch_bounds__(256) void proj_kernel(
    const float* __restrict__ X, const float* __restrict__ W,
    const float* __restrict__ bias, _Float16* __restrict__ Y) {
  __shared__ _Float16 As[64][34];
  __shared__ _Float16 Bs[32][66];
  __shared__ float    Cs[64][65];
  const int tid = threadIdx.x, wave = tid >> 5, lane = tid & 31;
  const int m0 = blockIdx.y * 64, n0 = blockIdx.x * 64;
  const int mt = (wave >> 2) * 32, nt = (wave & 3) * 16;
  const int ar = tid >> 2, ac = (tid & 3) * 8;  // A: 64 rows x 32 cols
  const int br = tid >> 3, bc = (tid & 7) * 8;  // B: 32 rows x 64 cols
  v8f acc0 = {}, acc1 = {};

  for (int k0 = 0; k0 < D_MODEL; k0 += 32) {
    const float* asrc = X + (size_t)(m0 + ar) * D_MODEL + k0 + ac;
    const float* bsrc = W + (size_t)(k0 + br) * D_MODEL + n0 + bc;
    __builtin_prefetch(asrc + 32, 0, 1);
    __builtin_prefetch(bsrc + 32 * D_MODEL, 0, 1);
#pragma unroll
    for (int j = 0; j < 8; ++j) As[ar][ac + j] = (_Float16)asrc[j];
#pragma unroll
    for (int j = 0; j < 8; ++j) Bs[br][bc + j] = (_Float16)bsrc[j];
    __syncthreads();
    v16h a0, a1, b;
    load_b_frag(&Bs[0][0], 66, nt, lane, b);
    load_a_frag(&As[0][0], 34, mt, lane, a0);
    load_a_frag(&As[0][0], 34, mt + 16, lane, a1);
    acc0 = wmma_f16(a0, b, acc0);
    acc1 = wmma_f16(a1, b, acc1);
    __syncthreads();
  }

  stage_c(Cs, mt, nt, lane, acc0, acc1);
  __syncthreads();
  // Coalesced epilogue: tile column range [n0, n0+64) lies in exactly one
  // head h = n0/64, with d = col contiguous -> contiguous f16 rows.
  const int row = tid >> 2;        // 0..63
  const int col = (tid & 3) * 16;  // 0,16,32,48
  const int m  = m0 + row;
  const int bb = m >> 11, ss = m & (SEQ - 1);
  const int h  = n0 >> 6;
  _Float16* dst =
      Y + (((size_t)(bb * NUM_HEADS + h) * SEQ) + ss) * DEPTH + col;
#pragma unroll
  for (int j = 0; j < 16; ++j) {
    dst[j] = (_Float16)(Cs[row][col + j] + bias[n0 + col + j]);
  }
}

// ---------------------------------------------------------------------------
// Kernel 2: logits[bh,q,k] = (qh[q,:] . kh[k,:]) / 8  + causal mask (-1e9)
// qh/kh f16 [BH, S, 64]. GEMM K-dim = DEPTH = 64 (2 WMMA K-steps).
// Fully-masked tiles (n0 >= m0+64) skip the matmul and stream -1e9.
// ---------------------------------------------------------------------------
__global__ __launch_bounds__(256) void logits_kernel(
    const _Float16* __restrict__ qh, const _Float16* __restrict__ kh,
    float* __restrict__ attn) {
  __shared__ _Float16 As[64][34];
  __shared__ _Float16 Bs[32][66];
  __shared__ float    Cs[64][65];
  const int tid = threadIdx.x, wave = tid >> 5, lane = tid & 31;
  const int bh = blockIdx.z;
  const int m0 = blockIdx.y * 64;  // query block
  const int n0 = blockIdx.x * 64;  // key block

  const int row = tid >> 2;        // 0..63
  const int col = (tid & 3) * 16;  // 0,16,32,48
  float* drow = attn + ((size_t)bh * SEQ + (m0 + row)) * SEQ + n0 + col;

  if (n0 >= m0 + 64) {  // tile entirely above the diagonal: all masked
#pragma unroll
    for (int j = 0; j < 16; ++j) drow[j] = -1.0e9f;
    return;
  }

  const _Float16* Q = qh + (size_t)bh * SEQ * DEPTH;
  const _Float16* K = kh + (size_t)bh * SEQ * DEPTH;
  const int mt = (wave >> 2) * 32, nt = (wave & 3) * 16;
  const int ar = tid >> 2, ac = (tid & 3) * 8;  // 64 x 32 loaders
  v8f acc0 = {}, acc1 = {};

#pragma unroll
  for (int k0 = 0; k0 < DEPTH; k0 += 32) {
    const _Float16* asrc = Q + (size_t)(m0 + ar) * DEPTH + k0 + ac;
    const _Float16* bsrc = K + (size_t)(n0 + ar) * DEPTH + k0 + ac;
#pragma unroll
    for (int j = 0; j < 8; ++j) As[ar][ac + j] = asrc[j];
    // Transpose on LDS store: Bs[depth][keypos]
#pragma unroll
    for (int j = 0; j < 8; ++j) Bs[ac + j][ar] = bsrc[j];
    __syncthreads();
    v16h a0, a1, b;
    load_b_frag(&Bs[0][0], 66, nt, lane, b);
    load_a_frag(&As[0][0], 34, mt, lane, a0);
    load_a_frag(&As[0][0], 34, mt + 16, lane, a1);
    acc0 = wmma_f16(a0, b, acc0);
    acc1 = wmma_f16(a1, b, acc1);
    __syncthreads();
  }

  stage_c(Cs, mt, nt, lane, acc0, acc1);
  __syncthreads();
  // Coalesced masked store: 16 consecutive f32 per thread (64B lanes).
  const int m = m0 + row;
#pragma unroll
  for (int j = 0; j < 16; ++j) {
    const int n = n0 + col + j;
    drow[j] = Cs[row][col + j] * 0.125f + (n > m ? -1.0e9f : 0.0f);
  }
}

// ---------------------------------------------------------------------------
// Kernel 3: in-place row softmax over S=2048. One 256-thread block per row,
// 8 values per thread held in registers (single global read + write).
// ---------------------------------------------------------------------------
__global__ __launch_bounds__(256) void softmax_kernel(float* __restrict__ attn) {
  float* p = attn + (size_t)blockIdx.x * SEQ;
  const int tid = threadIdx.x;
  __shared__ float red[8];

  float v[8];
  float lmax = -3.4e38f;
#pragma unroll
  for (int j = 0; j < 8; ++j) {
    v[j] = p[tid + j * 256];
    lmax = fmaxf(lmax, v[j]);
  }
#pragma unroll
  for (int off = 16; off > 0; off >>= 1)
    lmax = fmaxf(lmax, __shfl_down(lmax, off, 32));
  if ((tid & 31) == 0) red[tid >> 5] = lmax;
  __syncthreads();
  if (tid == 0) {
    float m = red[0];
    for (int i = 1; i < 8; ++i) m = fmaxf(m, red[i]);
    red[0] = m;
  }
  __syncthreads();
  const float rmax = red[0];
  __syncthreads();  // everyone has read red[0] before it is reused

  float lsum = 0.0f;
#pragma unroll
  for (int j = 0; j < 8; ++j) {
    v[j] = __expf(v[j] - rmax);
    lsum += v[j];
  }
#pragma unroll
  for (int off = 16; off > 0; off >>= 1) lsum += __shfl_down(lsum, off, 32);
  if ((tid & 31) == 0) red[tid >> 5] = lsum;
  __syncthreads();
  if (tid == 0) {
    float s = red[0];
    for (int i = 1; i < 8; ++i) s += red[i];
    red[0] = 1.0f / s;
  }
  __syncthreads();
  const float inv = red[0];
#pragma unroll
  for (int j = 0; j < 8; ++j) p[tid + j * 256] = v[j] * inv;
}

// ---------------------------------------------------------------------------
// Kernel 4: ctx = attn @ vh, written into concat layout [B, S, H*D] as f16.
// attn f32 [BH,S,S] (converted to f16 in LDS), vh f16 [BH,S,64]. N = 64.
// ---------------------------------------------------------------------------
__global__ __launch_bounds__(256) void ctx_kernel(
    const float* __restrict__ attn, const _Float16* __restrict__ vh,
    _Float16* __restrict__ concat) {
  __shared__ _Float16 As[64][34];
  __shared__ _Float16 Bs[32][66];
  __shared__ float    Cs[64][65];
  const int tid = threadIdx.x, wave = tid >> 5, lane = tid & 31;
  const int bh = blockIdx.z;
  const int m0 = blockIdx.y * 64;
  const float*    A = attn + ((size_t)bh * SEQ + m0) * SEQ;
  const _Float16* V = vh + (size_t)bh * SEQ * DEPTH;
  const int mt = (wave >> 2) * 32, nt = (wave & 3) * 16;
  const int ar = tid >> 2, ac = (tid & 3) * 8;
  const int br = tid >> 3, bc = (tid & 7) * 8;
  v8f acc0 = {}, acc1 = {};

  for (int k0 = 0; k0 < SEQ; k0 += 32) {
    const float*    asrc = A + (size_t)ar * SEQ + k0 + ac;
    const _Float16* bsrc = V + (size_t)(k0 + br) * DEPTH + bc;
    __builtin_prefetch(asrc + 32, 0, 1);
#pragma unroll
    for (int j = 0; j < 8; ++j) As[ar][ac + j] = (_Float16)asrc[j];
#pragma unroll
    for (int j = 0; j < 8; ++j) Bs[br][bc + j] = bsrc[j];
    __syncthreads();
    v16h a0, a1, b;
    load_b_frag(&Bs[0][0], 66, nt, lane, b);
    load_a_frag(&As[0][0], 34, mt, lane, a0);
    load_a_frag(&As[0][0], 34, mt + 16, lane, a1);
    acc0 = wmma_f16(a0, b, acc0);
    acc1 = wmma_f16(a1, b, acc1);
    __syncthreads();
  }

  stage_c(Cs, mt, nt, lane, acc0, acc1);
  __syncthreads();
  const int row = tid >> 2;        // 0..63 (query pos in tile)
  const int col = (tid & 3) * 16;  // depth chunk
  const int h = bh & (NUM_HEADS - 1), bb = bh >> 4;
  const int s = m0 + row;
  _Float16* dst =
      concat + ((size_t)bb * SEQ + s) * D_MODEL + h * DEPTH + col;
#pragma unroll
  for (int j = 0; j < 16; ++j) dst[j] = (_Float16)Cs[row][col + j];
}

// ---------------------------------------------------------------------------
// Kernel 5: out = concat @ wo + bo, f32 output, row-major [B*S, 1024].
// ---------------------------------------------------------------------------
__global__ __launch_bounds__(256) void out_kernel(
    const _Float16* __restrict__ Xh, const float* __restrict__ W,
    const float* __restrict__ bias, float* __restrict__ Y) {
  __shared__ _Float16 As[64][34];
  __shared__ _Float16 Bs[32][66];
  __shared__ float    Cs[64][65];
  const int tid = threadIdx.x, wave = tid >> 5, lane = tid & 31;
  const int m0 = blockIdx.y * 64, n0 = blockIdx.x * 64;
  const int mt = (wave >> 2) * 32, nt = (wave & 3) * 16;
  const int ar = tid >> 2, ac = (tid & 3) * 8;
  const int br = tid >> 3, bc = (tid & 7) * 8;
  v8f acc0 = {}, acc1 = {};

  for (int k0 = 0; k0 < D_MODEL; k0 += 32) {
    const _Float16* asrc = Xh + (size_t)(m0 + ar) * D_MODEL + k0 + ac;
    const float*    bsrc = W + (size_t)(k0 + br) * D_MODEL + n0 + bc;
    __builtin_prefetch(bsrc + 32 * D_MODEL, 0, 1);
#pragma unroll
    for (int j = 0; j < 8; ++j) As[ar][ac + j] = asrc[j];
#pragma unroll
    for (int j = 0; j < 8; ++j) Bs[br][bc + j] = (_Float16)bsrc[j];
    __syncthreads();
    v16h a0, a1, b;
    load_b_frag(&Bs[0][0], 66, nt, lane, b);
    load_a_frag(&As[0][0], 34, mt, lane, a0);
    load_a_frag(&As[0][0], 34, mt + 16, lane, a1);
    acc0 = wmma_f16(a0, b, acc0);
    acc1 = wmma_f16(a1, b, acc1);
    __syncthreads();
  }

  stage_c(Cs, mt, nt, lane, acc0, acc1);
  __syncthreads();
  const int row = tid >> 2;
  const int col = (tid & 3) * 16;
  float* dst = Y + (size_t)(m0 + row) * D_MODEL + n0 + col;
#pragma unroll
  for (int j = 0; j < 16; ++j) {
    dst[j] = Cs[row][col + j] + bias[n0 + col + j];
  }
}

// ---------------------------------------------------------------------------
extern "C" void kernel_launch(void* const* d_in, const int* in_sizes, int n_in,
                              void* d_out, int out_size, void* d_ws,
                              size_t ws_size, hipStream_t stream) {
  (void)in_sizes; (void)n_in; (void)out_size; (void)ws_size;
  const float* q  = (const float*)d_in[0];
  const float* k  = (const float*)d_in[1];
  const float* v  = (const float*)d_in[2];
  // d_in[3] = mask (triu, k=1) — applied analytically as the causal condition.
  const float* wq = (const float*)d_in[4];
  const float* bq = (const float*)d_in[5];
  const float* wk = (const float*)d_in[6];
  const float* bk = (const float*)d_in[7];
  const float* wv = (const float*)d_in[8];
  const float* bv = (const float*)d_in[9];
  const float* wo = (const float*)d_in[10];
  const float* bo = (const float*)d_in[11];

  float* out  = (float*)d_out;                       // [B, S, D_MODEL]
  float* attn = out + (size_t)BATCH * SEQ * D_MODEL; // [B, H, S, S]

  const size_t headElems = (size_t)BATCH * NUM_HEADS * SEQ * DEPTH;
  _Float16* qh     = (_Float16*)d_ws;
  _Float16* kh     = qh + headElems;
  _Float16* vh     = kh + headElems;
  _Float16* concat = vh + headElems;  // 4 x 8 MB = 32 MB workspace

  const dim3 blk(256);
  const dim3 gProj(D_MODEL / 64, (BATCH * SEQ) / 64);
  proj_kernel<<<gProj, blk, 0, stream>>>(q, wq, bq, qh);
  proj_kernel<<<gProj, blk, 0, stream>>>(k, wk, bk, kh);
  proj_kernel<<<gProj, blk, 0, stream>>>(v, wv, bv, vh);

  const dim3 gLog(SEQ / 64, SEQ / 64, BATCH * NUM_HEADS);
  logits_kernel<<<gLog, blk, 0, stream>>>(qh, kh, attn);

  softmax_kernel<<<dim3(BATCH * NUM_HEADS * SEQ), blk, 0, stream>>>(attn);

  const dim3 gCtx(1, SEQ / 64, BATCH * NUM_HEADS);
  ctx_kernel<<<gCtx, blk, 0, stream>>>(attn, vh, concat);

  out_kernel<<<gProj, blk, 0, stream>>>(concat, wo, bo, out);
}
